// MLPEncoder_83416854823500
// MI455X (gfx1250) — matrix operand order repaired and checked by reading, not compiled, not run on hardware
//
#include <hip/hip_runtime.h>

#define USE_TDM   1   // Tensor Data Mover tile loads (TENSORcnt)
#define USE_ASYNC 1   // fallback: per-lane global_load_async_to_lds (ASYNCcnt)

typedef __bf16 bf16_t;
typedef __attribute__((ext_vector_type(16))) __bf16 v16bf;
typedef __attribute__((ext_vector_type(8)))  __bf16 v8bf;
typedef __attribute__((ext_vector_type(4)))  __bf16 v4bf;
typedef __attribute__((ext_vector_type(8)))  float   v8f;
typedef __attribute__((ext_vector_type(4)))  float   f4;
typedef __attribute__((ext_vector_type(4)))  unsigned int v4u;
typedef __attribute__((ext_vector_type(4)))  unsigned int su4;
typedef __attribute__((ext_vector_type(8)))  unsigned int su8;

#define BM 256
#define BN 128
#define BK 32
#define LDA 40        // LDS row stride: 64B data + 16B pad (matches TDM pad config)
#define NTHREADS 512  // 16 waves: 4 (M) x 4 (N)

// generic pointer to LDS: low 32 bits are the LDS byte offset (ISA 10.2 aperture rule)
__device__ __forceinline__ unsigned lds_off(const void* p) {
    return (unsigned)(unsigned long long)p;
}

#if USE_TDM
// One-instruction DMA of a (tileRows x 32) bf16 tile (row stride K elems) into
// LDS with 16B padding per 64B row -> LDS row stride = 80B = LDA elements.
// 2D tensor: D# groups 2/3 are NULL -> 2-operand tensor_load_to_lds form.
__device__ __forceinline__ void tdm_load_2d(unsigned ldsOff, const void* gaddr,
                                            unsigned tileRows, unsigned Kelems)
{
    unsigned long long ga = (unsigned long long)gaddr;
    su4 g0;
    g0[0] = 1u;                                   // count=1 (valid user descriptor)
    g0[1] = ldsOff;                               // lds_addr (bytes)
    g0[2] = (unsigned)ga;                         // global_addr[31:0]
    g0[3] = (unsigned)(ga >> 32) | (2u << 30);    // global_addr[56:32] | type=2
    su8 g1;
    g1[0] = (1u << 16)                            // data_size = 1 -> 2 bytes
          | (1u << 20)                            // pad_enable
          | (3u << 22)                            // pad_interval: 16 DWORDs (64B)
          | (3u << 25);                           // pad_amount: 4 DWORDs (16B)
    g1[1] = (32u << 16);                          // tensor_dim0 = 32 (tile-exact)
    g1[2] = (tileRows & 0xffffu) << 16;           // tensor_dim1[15:0] = rows
    g1[3] = (32u << 16);                          // tile_dim0 = 32
    g1[4] = (tileRows & 0xffffu);                 // tile_dim1 = rows, tile_dim2 = 0
    g1[5] = Kelems;                               // tensor_dim0_stride[31:0]
    g1[6] = 0u;
    g1[7] = 0u;
    asm volatile("tensor_load_to_lds %0, %1" :: "s"(g0), "s"(g1) : "memory");
}
#endif

#if USE_ASYNC
__device__ __forceinline__ void async_ld_b128(unsigned ldsOff, const void* g) {
    asm volatile("global_load_async_to_lds_b128 %0, %1, off"
                 :: "v"(ldsOff), "v"((unsigned long long)g) : "memory");
}
#endif

// ---------------------------------------------------------------------------
// out = relu(A(rows x K) * W(dout x K)^T + bias) over compacted rows.
// Blocks whose row tile is beyond the (padded) compacted count exit early.
// FINAL=0: write bf16 hidden (chunk-local). FINAL=1: scatter fp32 to d_out
// via rowIdx (padding rows duplicate row idx[0] -> identical values, benign).
// ---------------------------------------------------------------------------
template<int FINAL>
__global__ __launch_bounds__(NTHREADS)
void mlp_gemm_relu(const bf16_t* __restrict__ A,
                   const bf16_t* __restrict__ W,
                   const float*  __restrict__ bias,
                   bf16_t* __restrict__ outH,
                   float*  __restrict__ outF,
                   const int* __restrict__ rowIdx,
                   const unsigned* __restrict__ cntPtr,
                   int rowStart, int K, int dout)
{
    const unsigned cp = (*cntPtr + (BM - 1u)) & ~(unsigned)(BM - 1u);
    const int rowBase = blockIdx.x * BM;                 // chunk-local
    if ((unsigned)(rowStart + rowBase) >= cp) return;    // uniform exit

    __shared__ __align__(32) bf16_t As[2][BM * LDA];
    __shared__ __align__(32) bf16_t Bs[2][BN * LDA];

    const int tid  = threadIdx.x;
    const int lane = tid & 31;
    const int wid  = tid >> 5;   // 0..15
    const int wM   = wid >> 2;   // 0..3  -> 64-row slab
    const int wN   = wid & 3;    // 0..3  -> 32-col slab
    const int hlf  = lane >> 4;
    const int l16  = lane & 15;
    const int colBase = blockIdx.y * BN;

    const bf16_t* Ab = A + (size_t)rowBase * K;
    const bf16_t* Wb = W + (size_t)colBase * K;

    v8f acc[4][2];
#pragma unroll
    for (int i = 0; i < 4; ++i)
#pragma unroll
        for (int j = 0; j < 2; ++j)
            acc[i][j] = (v8f)0.0f;

    auto compute = [&](int buf) {
        union { v16bf v; v8bf h[2]; } bfrag[2];
#pragma unroll
        for (int nt = 0; nt < 2; ++nt) {
            const bf16_t* p = &Bs[buf][(wN * 32 + nt * 16 + l16) * LDA + hlf * 16];
            bfrag[nt].h[0] = *(const v8bf*)(p);
            bfrag[nt].h[1] = *(const v8bf*)(p + 8);
        }
#pragma unroll
        for (int mt = 0; mt < 4; ++mt) {
            const int m = wM * 64 + mt * 16 + l16;
            union { v16bf v; v8bf h[2]; } af;
            af.h[0] = *(const v8bf*)(&As[buf][m * LDA + hlf * 16]);
            af.h[1] = *(const v8bf*)(&As[buf][m * LDA + hlf * 16 + 8]);
#pragma unroll
            for (int nt = 0; nt < 2; ++nt) {
                acc[mt][nt] = __builtin_amdgcn_wmma_f32_16x16x32_bf16(
                    false, af.v, false, bfrag[nt].v,
                    (short)0, acc[mt][nt], false, false);
            }
        }
    };

    const int nK = K / BK;

#if USE_TDM
    const unsigned ldsA[2] = { lds_off(&As[0][0]), lds_off(&As[1][0]) };
    const unsigned ldsB[2] = { lds_off(&Bs[0][0]), lds_off(&Bs[1][0]) };
    auto issue = [&](int kt, int buf) {
        tdm_load_2d(ldsA[buf], Ab + (size_t)kt * BK, BM, (unsigned)K);
        tdm_load_2d(ldsB[buf], Wb + (size_t)kt * BK, BN, (unsigned)K);
    };
    if (wid == 0) issue(0, 0);                   // wave 0 drives the TDM
    int buf = 0;
    for (int kt = 0; kt < nK; ++kt) {
        if (wid == 0) {
            if (kt + 1 < nK) {
                issue(kt + 1, buf ^ 1);          // overlap next tile
                __builtin_amdgcn_s_wait_tensorcnt(2);  // tile kt's 2 ops done
            } else {
                __builtin_amdgcn_s_wait_tensorcnt(0);
            }
        }
        __syncthreads();
        compute(buf);
        __syncthreads();   // all reads done before TDM overwrites this buffer
        buf ^= 1;
    }
#else
    // staging map: 16B chunks. A: 256x32 -> 1024 chunks (2/thread); B: 512 (1/thread)
    const int cA1 = tid + NTHREADS;
    const int rA0 = tid >> 2, kA0 = (tid & 3) * 8;
    const int rA1 = cA1 >> 2, kA1 = (cA1 & 3) * 8;
    const int rB  = tid >> 2, kB  = (tid & 3) * 8;
#if USE_ASYNC
    auto issue = [&](int kt, int buf) {
        const bf16_t* An = Ab + (size_t)kt * BK;
        const bf16_t* Wn = Wb + (size_t)kt * BK;
        async_ld_b128(lds_off(&As[buf][rA0 * LDA + kA0]), An + (size_t)rA0 * K + kA0);
        async_ld_b128(lds_off(&As[buf][rA1 * LDA + kA1]), An + (size_t)rA1 * K + kA1);
        async_ld_b128(lds_off(&Bs[buf][rB  * LDA + kB ]), Wn + (size_t)rB  * K + kB);
    };
    issue(0, 0);
    int buf = 0;
    for (int kt = 0; kt < nK; ++kt) {
        if (kt + 1 < nK) {
            issue(kt + 1, buf ^ 1);
            asm volatile("s_wait_asynccnt 0x3" ::: "memory");
        } else {
            asm volatile("s_wait_asynccnt 0x0" ::: "memory");
        }
        __syncthreads();
        compute(buf);
        __syncthreads();
        buf ^= 1;
    }
#else
    v4u a0 = *(const v4u*)(Ab + (size_t)rA0 * K + kA0);
    v4u a1 = *(const v4u*)(Ab + (size_t)rA1 * K + kA1);
    v4u b0 = *(const v4u*)(Wb + (size_t)rB  * K + kB);
    for (int kt = 0; kt < nK; ++kt) {
        __syncthreads();
        *(v4u*)(&As[0][rA0 * LDA + kA0]) = a0;
        *(v4u*)(&As[0][rA1 * LDA + kA1]) = a1;
        *(v4u*)(&Bs[0][rB  * LDA + kB ]) = b0;
        __syncthreads();
        if (kt + 1 < nK) {
            const bf16_t* An = Ab + (size_t)(kt + 1) * BK;
            const bf16_t* Wn = Wb + (size_t)(kt + 1) * BK;
            a0 = *(const v4u*)(An + (size_t)rA0 * K + kA0);
            a1 = *(const v4u*)(An + (size_t)rA1 * K + kA1);
            b0 = *(const v4u*)(Wn + (size_t)rB  * K + kB);
        }
        compute(0);
    }
#endif
#endif

    // epilogue: bias + relu. C layout: VGPR j -> M = j + 8*hlf, N = l16
#pragma unroll
    for (int nt = 0; nt < 2; ++nt) {
        const int colG = colBase + wN * 32 + nt * 16 + l16;
        const float bv = bias[colG];
#pragma unroll
        for (int mt = 0; mt < 4; ++mt) {
#pragma unroll
            for (int j = 0; j < 8; ++j) {
                float v = acc[mt][nt][j] + bv;
                v = v > 0.0f ? v : 0.0f;
                const int rowG = rowBase + wM * 64 + mt * 16 + j + hlf * 8;
                if (FINAL) {
                    const int realRow = rowIdx[rowStart + rowG];
                    outF[(size_t)realRow * 256 + colG] = v;
                } else {
                    outH[(size_t)rowG * dout + colG] = (bf16_t)v;
                }
            }
        }
    }
}

__global__ void zero_counters(unsigned* __restrict__ cnt)
{
    if (threadIdx.x < 2) cnt[threadIdx.x] = 0;
}

// one atomic per wave per sign; within-wave order via ballot prefix
__global__ void compact_rows(const int* __restrict__ obs, int N,
                             int* __restrict__ idxNeg, int* __restrict__ idxPos,
                             unsigned* __restrict__ cnt)
{
    int i = blockIdx.x * blockDim.x + threadIdx.x;
    int lane = threadIdx.x & 31;
    int o = (i < N) ? obs[i] : 0;
    unsigned long long mneg = __ballot(o == -1);
    unsigned long long mpos = __ballot(o == 1);
    unsigned long long pre = (1ull << lane) - 1ull;
    unsigned bneg = 0, bpos = 0;
    if (lane == 0) {
        if (mneg) bneg = atomicAdd(&cnt[0], (unsigned)__popcll(mneg));
        if (mpos) bpos = atomicAdd(&cnt[1], (unsigned)__popcll(mpos));
    }
    bneg = __shfl(bneg, 0, 32);
    bpos = __shfl(bpos, 0, 32);
    if (o == -1)      idxNeg[bneg + __popcll(mneg & pre)] = i;
    else if (o == 1)  idxPos[bpos + __popcll(mpos & pre)] = i;
}

// pad index list to a BM multiple with a copy of idx[0] (deterministic)
__global__ void pad_indices(int* __restrict__ idx, const unsigned* __restrict__ cnt)
{
    unsigned c = *cnt;
    unsigned cp = (c + (BM - 1u)) & ~(unsigned)(BM - 1u);
    unsigned i = c + threadIdx.x;
    if (i < cp) idx[i] = (c > 0) ? idx[0] : 0;
}

__global__ void convert_f32_bf16_v4(const float* __restrict__ in,
                                    bf16_t* __restrict__ out, int n4)
{
    int i = blockIdx.x * blockDim.x + threadIdx.x;
    if (i >= n4) return;
    f4 v = *(const f4*)(in + (size_t)i * 4);
    v4bf o;
    o.x = (bf16_t)v.x; o.y = (bf16_t)v.y; o.z = (bf16_t)v.z; o.w = (bf16_t)v.w;
    *(v4bf*)(out + (size_t)i * 4) = o;
}

// gather compacted rows of x into packed bf16 chunk buffer
__global__ void gather_convert(const float* __restrict__ x,
                               const int* __restrict__ idx,
                               const unsigned* __restrict__ cnt,
                               int rowStart, bf16_t* __restrict__ out)
{
    unsigned cp = (*cnt + (BM - 1u)) & ~(unsigned)(BM - 1u);
    int t = blockIdx.x * blockDim.x + threadIdx.x;
    int lr = t >> 6;                 // chunk-local row (64 threads/row, 4 elems each)
    int gr = rowStart + lr;
    if ((unsigned)gr >= cp) return;
    int src = idx[gr];
    int col = (t & 63) << 2;
    f4 v = *(const f4*)(x + (size_t)src * 256 + col);
    v4bf o;
    o.x = (bf16_t)v.x; o.y = (bf16_t)v.y; o.z = (bf16_t)v.z; o.w = (bf16_t)v.w;
    *(v4bf*)(out + (size_t)lr * 256 + col) = o;
}

// obs==0 rows: x passes through exactly (fp32)
__global__ void select_passthrough(const float* __restrict__ x,
                                   const int* __restrict__ obs,
                                   float* __restrict__ out)
{
    int idx = blockIdx.x * blockDim.x + threadIdx.x;
    int row = idx >> 6;
    int col = (idx & 63) << 2;
    if (obs[row] == 0) {
        size_t o = (size_t)row * 256 + col;
        *(f4*)(out + o) = *(const f4*)(x + o);
    }
}

extern "C" void kernel_launch(void* const* d_in, const int* in_sizes, int n_in,
                              void* d_out, int out_size, void* d_ws, size_t ws_size,
                              hipStream_t stream)
{
    const float* x   = (const float*)d_in[0];
    const int*   obs = (const int*)d_in[1];
    const float* Wf[6];
    const float* bf_[6];
    for (int i = 0; i < 6; ++i) {
        Wf[i]  = (const float*)d_in[2 + 2 * i];
        bf_[i] = (const float*)d_in[3 + 2 * i];
    }
    const int N = in_sizes[1];
    float* out = (float*)d_out;

    const int wsizes[6] = {1024 * 256, 1024 * 1024, 256 * 1024,
                           1024 * 256, 1024 * 1024, 256 * 1024};
    size_t woff[6], wtot = 0;
    for (int i = 0; i < 6; ++i) { woff[i] = wtot; wtot += (size_t)wsizes[i]; }

    // ws layout: [bf16 weights][idxNeg: N][idxPos: N][cnt: 4 u32][xb][h1][h2]
    char* ws = (char*)d_ws;
    bf16_t*   wb     = (bf16_t*)ws;
    int*      idxNeg = (int*)(ws + wtot * 2);
    int*      idxPos = idxNeg + N;
    unsigned* cnt    = (unsigned*)(idxPos + N);
    bf16_t*   xb     = (bf16_t*)((char*)cnt + 16);

    size_t used  = wtot * 2 + (size_t)N * 8 + 16;
    size_t avail = (ws_size > used) ? ws_size - used : 0;
    long ch = (long)(avail / 4608);          // (256+1024+1024)*2 bytes per row
    int CH = (int)((ch / BM) * BM);
    if (CH < BM)    CH = BM;
    if (CH > 32768) CH = 32768;              // keep h1/h2 L2-resident

    bf16_t* h1 = xb + (size_t)CH * 256;
    bf16_t* h2 = h1 + (size_t)CH * 1024;

    // compaction (counts stay on device; grids are worst-case + early-exit)
    zero_counters<<<1, 32, 0, stream>>>(cnt);
    compact_rows<<<(N + 255) / 256, 256, 0, stream>>>(obs, N, idxNeg, idxPos, cnt);
    pad_indices<<<1, BM, 0, stream>>>(idxNeg, cnt + 0);
    pad_indices<<<1, BM, 0, stream>>>(idxPos, cnt + 1);

    // weights fp32 -> bf16
    for (int i = 0; i < 6; ++i) {
        int n4 = wsizes[i] / 4;
        convert_f32_bf16_v4<<<(n4 + 255) / 256, 256, 0, stream>>>(
            Wf[i], wb + woff[i], n4);
    }

    // obs==0 rows
    select_passthrough<<<(N * 64) / 256, 256, 0, stream>>>(x, obs, out);

    const int gX = CH / BM;
    for (int m = 0; m < 2; ++m) {            // 0 = neg (obs==-1), 1 = pos (obs==1)
        const int* idx = m ? idxPos : idxNeg;
        const unsigned* c = cnt + m;
        for (int rs = 0; rs < N; rs += CH) {
            gather_convert<<<CH / 4, 256, 0, stream>>>(x, idx, c, rs, xb);
            mlp_gemm_relu<0><<<dim3(gX, 1024 / BN), NTHREADS, 0, stream>>>(
                xb, wb + woff[3 * m + 0], bf_[3 * m + 0],
                h1, nullptr, nullptr, c, rs, 256, 1024);
            mlp_gemm_relu<0><<<dim3(gX, 1024 / BN), NTHREADS, 0, stream>>>(
                h1, wb + woff[3 * m + 1], bf_[3 * m + 1],
                h2, nullptr, nullptr, c, rs, 1024, 1024);
            mlp_gemm_relu<1><<<dim3(gX, 256 / BN), NTHREADS, 0, stream>>>(
                h2, wb + woff[3 * m + 2], bf_[3 * m + 2],
                nullptr, out, idx, c, rs, 1024, 256);
        }
    }
}
